// RNN_23416161698483
// MI455X (gfx1250) — matrix operand order repaired and checked by reading, not compiled, not run on hardware
//
#include <hip/hip_runtime.h>
#include <hip/hip_bf16.h>
#include <math.h>

#define T_STEPS 1024
#define BATCH   64
#define NIN     128
#define NH      512
#define WREC_C  (NH + NIN + 1)   // 641
#define NWG_REC 4
#define REC_THREADS 256

typedef __attribute__((ext_vector_type(16))) __bf16 v16bf;
typedef __attribute__((ext_vector_type(8)))  float  v8f;

union Frag16 {
    v16bf v;
    uint4 q[2];
    unsigned short s[16];
    __bf16 h[16];
};

// hardware f32 -> bf16 convert (v_cvt_*_bf16_f32), not manual bit twiddling
__device__ __forceinline__ unsigned short f2bf(float f) {
    return __builtin_bit_cast(unsigned short, (__bf16)f);
}

// ---------------------------------------------------------------------------
// Kernel 0: split + pack w_rec (f32, 512x641) -> w_h bf16, w_x bf16, bias f32
// ---------------------------------------------------------------------------
__global__ __launch_bounds__(256) void pack_weights(
        const float* __restrict__ w_rec,
        unsigned short* __restrict__ wh,
        unsigned short* __restrict__ wx,
        float* __restrict__ bias) {
    int idx = blockIdx.x * blockDim.x + threadIdx.x;
    if (idx >= NH * WREC_C) return;
    int r = idx / WREC_C;
    int c = idx - r * WREC_C;
    float v = w_rec[idx];
    if (c < NH)            wh[r * NH + c] = f2bf(v);
    else if (c < NH + NIN) wx[r * NIN + (c - NH)] = f2bf(v);
    else                   bias[r] = v;
}

// ---------------------------------------------------------------------------
// Kernel 1: h0 f32 -> bf16 ping buffer, zero the grid-barrier counter
// ---------------------------------------------------------------------------
__global__ __launch_bounds__(256) void init_h(
        const float* __restrict__ h0,
        unsigned short* __restrict__ hbuf0,
        int* __restrict__ counter) {
    int idx = blockIdx.x * blockDim.x + threadIdx.x;
    if (idx < BATCH * NH) hbuf0[idx] = f2bf(h0[idx]);
    if (idx == 0) *counter = 0;
}

// ---------------------------------------------------------------------------
// Kernel 2: xin = x @ w_x^T + b  -> written into d_out (overwritten by scan)
// One 16x16 tile per wave; K=128 -> 4 x wmma_f32_16x16x32_bf16.
// ---------------------------------------------------------------------------
__global__ __launch_bounds__(256) void xin_gemm(
        const float* __restrict__ x,
        const unsigned short* __restrict__ wx,
        const float* __restrict__ bias,
        float* __restrict__ out) {
    const int lane = threadIdx.x & 31;
    const int wave = threadIdx.x >> 5;
    const int tile = blockIdx.x * 8 + wave;   // 131072 tiles total
    const int mt   = tile >> 5;               // 0..4095 (rows of T*B)
    const int nt   = tile & 31;               // 0..31   (cols of NH)
    const int r    = lane & 15;
    const int half = lane >> 4;

    v8f acc = {};
#pragma unroll
    for (int k0 = 0; k0 < 4; ++k0) {
        // A fragment: 16x32 bf16, lane(r,half) holds M=r,
        // K = {32k0+8*half+0..7} and {32k0+16+8*half+0..7}; convert f32->bf16.
        Frag16 a;
        const float* xa = x + (size_t)(mt * 16 + r) * NIN + k0 * 32 + half * 8;
        float tmp[16];
        *(float4*)(tmp + 0)  = *(const float4*)(xa);
        *(float4*)(tmp + 4)  = *(const float4*)(xa + 4);
        *(float4*)(tmp + 8)  = *(const float4*)(xa + 16);
        *(float4*)(tmp + 12) = *(const float4*)(xa + 20);
#pragma unroll
        for (int i = 0; i < 16; ++i) a.h[i] = (__bf16)tmp[i];

        // B fragment: 32x16 bf16, lane(r,half) holds column N=r,
        // K = {8*half+0..7} and {16+8*half+0..7} of w_x row (nt*16+r).
        Frag16 b;
        const unsigned short* wb =
            wx + (size_t)(nt * 16 + r) * NIN + k0 * 32 + half * 8;
        b.q[0] = *(const uint4*)(wb);
        b.q[1] = *(const uint4*)(wb + 16);

        acc = __builtin_amdgcn_wmma_f32_16x16x32_bf16(
            false, a.v, false, b.v, (short)0, acc, false, false);
    }

    const float bv = bias[nt * 16 + r];
#pragma unroll
    for (int v = 0; v < 8; ++v) {
        const int row = mt * 16 + v + 8 * half;       // C/D layout: M=v(+8*half)
        out[(size_t)row * NH + nt * 16 + r] = acc[v] + bv;
    }
}

// ---------------------------------------------------------------------------
// Kernel 3: persistent recurrent scan.
// 4 WGs x 256 threads; WG owns 128 columns of h. w_h slice (128KB bf16)
// resident in LDS for all steps; h_{t-1} staged global->LDS via TDM each step.
// ---------------------------------------------------------------------------
__global__ __launch_bounds__(REC_THREADS) void rnn_scan(
        const unsigned short* __restrict__ wh,
        unsigned short* __restrict__ hbuf0,
        unsigned short* __restrict__ hbuf1,
        float* __restrict__ out,
        int* __restrict__ counter) {
    extern __shared__ unsigned short smem[];
    unsigned short* w_lds = smem;              // 128*512 bf16 = 128 KB
    unsigned short* h_lds = smem + 128 * NH;   //  64*512 bf16 =  64 KB

    const int wg     = blockIdx.x;             // 0..3
    const int n_base = wg * 128;
    const int lane   = threadIdx.x & 31;
    const int wave   = threadIdx.x >> 5;       // 0..7
    const int mt     = wave >> 1;              // M-tile 0..3
    const int ng     = (wave & 1) * 4;         // N-tile group base (0 or 4)
    const int r      = lane & 15;
    const int half   = lane >> 4;

    // Load this WG's w_h slice into LDS once (stays resident for 1024 steps).
    {
        const uint4* src = (const uint4*)(wh + (size_t)n_base * NH);
        uint4* dst = (uint4*)w_lds;
        for (int i = threadIdx.x; i < 128 * NH / 8; i += blockDim.x)
            dst[i] = src[i];
    }
    __syncthreads();

    unsigned short* hb[2] = {hbuf0, hbuf1};

    for (int t = 0; t < T_STEPS; ++t) {
        const unsigned short* hprev = hb[t & 1];
        unsigned short*       hnext = hb[(t + 1) & 1];

        // ---- stage full h_{t-1} (64x512 bf16 = 64KB) into LDS via TDM ----
#if defined(__gfx1250__) && __has_builtin(__builtin_amdgcn_tensor_load_to_lds) && \
    __has_builtin(__builtin_amdgcn_s_wait_tensorcnt)
        if (wave == 0) {
            typedef unsigned int u32;
            typedef __attribute__((ext_vector_type(4))) u32 u32x4;
            typedef __attribute__((ext_vector_type(8))) int i32x8;
            typedef __attribute__((ext_vector_type(4))) int i32x4;
            const u32 lds_addr = (u32)(uintptr_t)(void*)h_lds;
            const unsigned long long ga = (unsigned long long)(uintptr_t)hprev;
            u32x4 g0 = {};
            g0[0] = 1u;                                   // count=1, user desc
            g0[1] = lds_addr;                             // lds_addr[31:0]
            g0[2] = (u32)(ga & 0xFFFFFFFFu);              // global_addr lo
            g0[3] = (u32)((ga >> 32) & 0x1FFFFFFu) | (2u << 30); // hi + type=2
            i32x8 g1 = {};
            g1[0] = (int)(1u << 16);          // data_size=1 (2B), no multicast
            g1[1] = (int)(512u << 16);        // tensor_dim0 = 512 (bits 79:48)
            g1[2] = (int)(64u << 16);         // tensor_dim1 = 64  (bits 111:80)
            g1[3] = (int)(512u << 16);        // tile_dim0   = 512 (bits 127:112)
            g1[4] = 64;                       // tile_dim1   = 64  (bits 143:128)
            g1[5] = 512;                      // tensor_dim0_stride = 512
            i32x4 g2 = {};
            i32x4 g3 = {};
            i32x8 g4 = {};
            __builtin_amdgcn_tensor_load_to_lds(g0, g1, g2, g3, g4, 0);
            __builtin_amdgcn_s_wait_tensorcnt(0);
        }
        __syncthreads();
#else
        {
            const uint4* src = (const uint4*)hprev;
            uint4* dst = (uint4*)h_lds;
            for (int i = threadIdx.x; i < BATCH * NH / 8; i += blockDim.x)
                dst[i] = src[i];
        }
        __syncthreads();
#endif

        // ---- 64x128 tile GEMM: 4 tiles per wave, K=512 -> 64 WMMAs/wave ----
        v8f acc[4] = {v8f{}, v8f{}, v8f{}, v8f{}};
        for (int k0 = 0; k0 < 16; ++k0) {
            Frag16 a;   // A 16x32 bf16 frag of h_prev rows mt*16..mt*16+15
            const unsigned short* ha =
                h_lds + (mt * 16 + r) * NH + k0 * 32 + half * 8;
            a.q[0] = *(const uint4*)(ha);
            a.q[1] = *(const uint4*)(ha + 16);
#pragma unroll
            for (int j = 0; j < 4; ++j) {
                Frag16 b;   // B 32x16 frag: K {8h+0..7},{16+8h+0..7} of w_h row
                const unsigned short* wb =
                    w_lds + ((ng + j) * 16 + r) * NH + k0 * 32 + half * 8;
                b.q[0] = *(const uint4*)(wb);
                b.q[1] = *(const uint4*)(wb + 16);
                acc[j] = __builtin_amdgcn_wmma_f32_16x16x32_bf16(
                    false, a.v, false, b.v, (short)0, acc[j], false, false);
            }
        }

        // ---- epilogue: h = tanh(acc + xin); out[t] <- h; hnext <- bf16(h) ----
#pragma unroll
        for (int j = 0; j < 4; ++j) {
            const int hcol = n_base + (ng + j) * 16 + r;
#pragma unroll
            for (int v = 0; v < 8; ++v) {
                const int brow = mt * 16 + v + 8 * half;
                const size_t off = (size_t)t * BATCH * NH +
                                   (size_t)brow * NH + hcol;
                const float val = tanhf(acc[j][v] + out[off]);
                out[off] = val;
                hnext[brow * NH + hcol] = f2bf(val);
            }
        }

        // ---- grid-wide barrier (monotonic counter, zeroed by init_h) ----
        __threadfence();
        __syncthreads();
        if (threadIdx.x == 0) {
            __hip_atomic_fetch_add(counter, 1, __ATOMIC_ACQ_REL,
                                   __HIP_MEMORY_SCOPE_AGENT);
            const int target = (t + 1) * NWG_REC;
            while (__hip_atomic_load(counter, __ATOMIC_ACQUIRE,
                                     __HIP_MEMORY_SCOPE_AGENT) < target) { }
        }
        __syncthreads();
    }
}

// ---------------------------------------------------------------------------
extern "C" void kernel_launch(void* const* d_in, const int* in_sizes, int n_in,
                              void* d_out, int out_size, void* d_ws, size_t ws_size,
                              hipStream_t stream) {
    const float* x     = (const float*)d_in[0];   // (T, B, N_IN) f32
    const float* h0    = (const float*)d_in[1];   // (B, N_H)     f32
    const float* w_rec = (const float*)d_in[2];   // (N_H, 641)   f32
    float* out = (float*)d_out;                   // (T, B, N_H)  f32

    char* ws = (char*)d_ws;
    unsigned short* wh    = (unsigned short*)(ws + 0);        // 512 KB bf16
    unsigned short* wx    = (unsigned short*)(ws + 524288);   // 128 KB bf16
    float*          bias  = (float*)(ws + 655360);            //   2 KB f32
    unsigned short* hbuf0 = (unsigned short*)(ws + 657408);   //  64 KB bf16
    unsigned short* hbuf1 = (unsigned short*)(ws + 722944);   //  64 KB bf16
    int*            ctr   = (int*)(ws + 788480);

    const int packN = NH * WREC_C;
    pack_weights<<<(packN + 255) / 256, 256, 0, stream>>>(w_rec, wh, wx, bias);
    init_h<<<(BATCH * NH + 255) / 256, 256, 0, stream>>>(h0, hbuf0, ctr);

    // 4096 m-tiles x 32 n-tiles = 131072 tiles, 8 waves per block
    xin_gemm<<<131072 / 8, 256, 0, stream>>>(x, wx, bias, out);

    const size_t lds_bytes = (size_t)(128 * NH + BATCH * NH) * sizeof(unsigned short); // 192 KB
    rnn_scan<<<NWG_REC, REC_THREADS, lds_bytes, stream>>>(wh, hbuf0, hbuf1, out, ctr);
}
